// Stagate_4337916969124
// MI455X (gfx1250) — matrix operand order repaired and checked by reading, not compiled, not run on hardware
//
#include <hip/hip_runtime.h>
#include <hip/hip_bf16.h>

typedef __attribute__((ext_vector_type(16))) __bf16 v16bf;
typedef __attribute__((ext_vector_type(8)))  float  v8f;

static constexpr int NN   = 100000;
static constexpr int EE   = 1600000;
static constexpr int IND  = 512;
static constexpr int HIDD = 256;
static constexpr int OUTD = 32;

__device__ __forceinline__ v8f wmma_bf16(v16bf a, v16bf b, v8f c) {
    return __builtin_amdgcn_wmma_f32_16x16x32_bf16(
        false, a, false, b, (short)0, c, false, false);
}

// ---------------------------------------------------------------------------
// GEMM: C[M,Nc](f32, optional) / C2[M,Nc](bf16, optional) = A * B
//   A  : [M][K]  bf16 row-major
//   Bt : [Nc][K] bf16 row-major  (i.e. B transposed -> per-lane fragment rows
//        are two separate 16B runs, loadable as two global_load_b128)
// Block: 256 threads = 8 waves. Wave tile: 32(M) x 32(N) -> 4 accumulators,
// A/B fragments each reused twice. Block tile: 256(M) x 32(N).
// No LDS, no barriers: fragments stream straight from global (B is L2-hot).
// ---------------------------------------------------------------------------
__global__ __launch_bounds__(256) void gemm_bf16_wmma(
    const __bf16* __restrict__ A, const __bf16* __restrict__ Bt,
    float* __restrict__ C, __bf16* __restrict__ C2,
    int M, int K, int Nc)
{
    const int tid  = threadIdx.x;
    const int wave = tid >> 5;
    const int lane = tid & 31;
    const int half = lane >> 4;          // 0/1
    const int lr   = lane & 15;
    const int rowBase = blockIdx.x * 256 + wave * 32;
    const int colBase = blockIdx.y * 32;

    // A fragment rows (16x32 ISA layout): lane<16 -> K {0..7,16..23},
    // lane>=16 -> K {8..15,24..31}; rows clamped so EXEC stays all-1.
    int r0 = rowBase + lr;       if (r0 >= M) r0 = M - 1;
    int r1 = rowBase + 16 + lr;  if (r1 >= M) r1 = M - 1;
    const __bf16* a0p = A + (size_t)r0 * K + 8 * half;
    const __bf16* a1p = A + (size_t)r1 * K + 8 * half;
    // B fragments (32x16): lanes 0-15 hold K {0..7,16..23}, 16-31 hold K {8..15,24..31};
    // column n = colBase + (cb*16) + lr -> two 16B runs of Bt.
    const __bf16* b0p = Bt + (size_t)(colBase + lr) * K + 8 * half;
    const __bf16* b1p = Bt + (size_t)(colBase + 16 + lr) * K + 8 * half;

    v8f acc00 = {}, acc01 = {}, acc10 = {}, acc11 = {};

    for (int k0 = 0; k0 < K; k0 += 32) {
        v16bf a0, a1, b0, b1;
        *((uint4*)&a0)     = *(const uint4*)(a0p + k0);
        *((uint4*)&a0 + 1) = *(const uint4*)(a0p + k0 + 16);
        *((uint4*)&a1)     = *(const uint4*)(a1p + k0);
        *((uint4*)&a1 + 1) = *(const uint4*)(a1p + k0 + 16);
        *((uint4*)&b0)     = *(const uint4*)(b0p + k0);
        *((uint4*)&b0 + 1) = *(const uint4*)(b0p + k0 + 16);
        *((uint4*)&b1)     = *(const uint4*)(b1p + k0);
        *((uint4*)&b1 + 1) = *(const uint4*)(b1p + k0 + 16);
        // prefetch next K-chunk of the streaming A operand
        __builtin_prefetch(a0p + k0 + 32, 0, 1);
        __builtin_prefetch(a1p + k0 + 32, 0, 1);
        acc00 = wmma_bf16(a0, b0, acc00);
        acc01 = wmma_bf16(a0, b1, acc01);
        acc10 = wmma_bf16(a1, b0, acc10);
        acc11 = wmma_bf16(a1, b1, acc11);
    }

    // Epilogue: VGPR g -> row (subtile + g + 8*half), col = colBase+lr (+16)
    const int c0 = colBase + lr;
    const int orow = rowBase + (half << 3);
    #pragma unroll
    for (int g = 0; g < 8; ++g) {
        int gr = orow + g;
        if (gr < M) {
            size_t o = (size_t)gr * Nc + c0;
            if (C)  { C[o] = acc00[g];          C[o + 16] = acc01[g]; }
            if (C2) { C2[o] = (__bf16)acc00[g]; C2[o + 16] = (__bf16)acc01[g]; }
        }
        int gr1 = gr + 16;
        if (gr1 < M) {
            size_t o = (size_t)gr1 * Nc + c0;
            if (C)  { C[o] = acc10[g];          C[o + 16] = acc11[g]; }
            if (C2) { C2[o] = (__bf16)acc10[g]; C2[o + 16] = (__bf16)acc11[g]; }
        }
    }
}

// ---------------------------------------------------------------------------
__global__ __launch_bounds__(256) void zero_f32(float* __restrict__ p, int n) {
    int i = blockIdx.x * 256 + threadIdx.x;
    if (i < n) p[i] = 0.f;
}

// streaming fp32 -> bf16 (8 elements / thread, b128 in, b128 out)
__global__ __launch_bounds__(256) void f32_to_bf16(
    const float* __restrict__ in, __bf16* __restrict__ out, int n8)
{
    int i = blockIdx.x * 256 + threadIdx.x;
    if (i >= n8) return;
    const float4* p = (const float4*)in + (size_t)2 * i;
    float4 v0 = p[0], v1 = p[1];
    union { __bf16 b[8]; uint4 u; } pk;
    pk.b[0] = (__bf16)v0.x; pk.b[1] = (__bf16)v0.y;
    pk.b[2] = (__bf16)v0.z; pk.b[3] = (__bf16)v0.w;
    pk.b[4] = (__bf16)v1.x; pk.b[5] = (__bf16)v1.y;
    pk.b[6] = (__bf16)v1.z; pk.b[7] = (__bf16)v1.w;
    ((uint4*)out)[i] = pk.u;
}

// streaming ELU(fp32) -> bf16
__global__ __launch_bounds__(256) void elu_to_bf16(
    const float* __restrict__ in, __bf16* __restrict__ out, int n8)
{
    int i = blockIdx.x * 256 + threadIdx.x;
    if (i >= n8) return;
    const float4* p = (const float4*)in + (size_t)2 * i;
    float4 v0 = p[0], v1 = p[1];
    float f[8] = { v0.x, v0.y, v0.z, v0.w, v1.x, v1.y, v1.z, v1.w };
    union { __bf16 b[8]; uint4 u; } pk;
    #pragma unroll
    for (int j = 0; j < 8; ++j) {
        float v = f[j];
        pk.b[j] = (__bf16)(v > 0.f ? v : (__expf(v) - 1.f));
    }
    ((uint4*)out)[i] = pk.u;
}

// W[K][Nc] fp32 -> Wt[Nc][K] bf16
__global__ __launch_bounds__(256) void transpose_to_bf16(
    const float* __restrict__ W, __bf16* __restrict__ Wt, int K, int Nc)
{
    int i = blockIdx.x * 256 + threadIdx.x;
    if (i >= K * Nc) return;
    int k = i / Nc, n = i - k * Nc;
    Wt[(size_t)n * K + k] = (__bf16)W[i];
}

// a_src[i] = xp1[i,:]·att_src ; a_dst[i] = xp1[i,:]·att_dst  (wave per node)
__global__ __launch_bounds__(256) void attn_logits(
    const float* __restrict__ xp, const float* __restrict__ wsrc,
    const float* __restrict__ wdst, float* __restrict__ a_src,
    float* __restrict__ a_dst, int n)
{
    int node = (blockIdx.x * 256 + threadIdx.x) >> 5;
    int lane = threadIdx.x & 31;
    if (node >= n) return;
    const float* row = xp + (size_t)node * HIDD;
    float s1 = 0.f, s2 = 0.f;
    #pragma unroll
    for (int j = 0; j < HIDD / 32; ++j) {
        int c = lane + 32 * j;
        float v = row[c];
        s1 += v * wsrc[c];
        s2 += v * wdst[c];
    }
    #pragma unroll
    for (int off = 16; off > 0; off >>= 1) {
        s1 += __shfl_down(s1, off, 32);
        s2 += __shfl_down(s2, off, 32);
    }
    if (lane == 0) { a_src[node] = s1; a_dst[node] = s2; }
}

// pass 1: alpha = sigmoid(...); segment-max via int atomicMax (alpha>0 so
// IEEE order == int order and 0-init is a valid identity)
__global__ __launch_bounds__(256) void edge_alpha(
    const int* __restrict__ src, const int* __restrict__ dst,
    const float* __restrict__ a_src, const float* __restrict__ a_dst,
    float* __restrict__ alpha, int* __restrict__ mbits, int E)
{
    int e = blockIdx.x * 256 + threadIdx.x;
    if (e >= E) return;
    float x  = a_src[src[e]] + a_dst[dst[e]];
    float al = 1.f / (1.f + __expf(-x));
    alpha[e] = al;
    atomicMax(&mbits[dst[e]], __float_as_int(al));
}

// pass 2: ex = exp(alpha - m[dst]); denom[dst] += ex  (ex overwrites alpha)
__global__ __launch_bounds__(256) void edge_exp(
    const int* __restrict__ dst, float* __restrict__ alpha,
    const int* __restrict__ mbits, float* __restrict__ denom, int E)
{
    int e = blockIdx.x * 256 + threadIdx.x;
    if (e >= E) return;
    int d = dst[e];
    float ex = __expf(alpha[e] - __int_as_float(mbits[d]));
    alpha[e] = ex;
    atomicAdd(&denom[d], ex);
}

// pass 3: out[dst,:] += xp[src,:] * ex/(denom[dst]+eps)   (wave per edge)
__global__ __launch_bounds__(256) void edge_scatter(
    const int* __restrict__ src, const int* __restrict__ dst,
    const float* __restrict__ ex, const float* __restrict__ denom,
    const float* __restrict__ xp, float* __restrict__ out, int E)
{
    int e    = (blockIdx.x * 256 + threadIdx.x) >> 5;
    int lane = threadIdx.x & 31;
    if (e >= E) return;
    int s = src[e], d = dst[e];
    float a = ex[e] / (denom[d] + 1e-16f);
    const float* xr = xp + (size_t)s * HIDD;
    float* orow     = out + (size_t)d * HIDD;
    #pragma unroll
    for (int j = 0; j < HIDD / 32; ++j) {
        int c = lane + 32 * j;                 // coalesced 128B per step
        atomicAdd(&orow[c], xr[c] * a);
    }
}

// ---------------------------------------------------------------------------
extern "C" void kernel_launch(void* const* d_in, const int* in_sizes, int n_in,
                              void* d_out, int out_size, void* d_ws, size_t ws_size,
                              hipStream_t stream)
{
    const float* features = (const float*)d_in[0];
    const int*   eidx     = (const int*)d_in[1];
    const float* W1       = (const float*)d_in[2];
    const float* W2       = (const float*)d_in[3];
    const float* att_src  = (const float*)d_in[4];
    const float* att_dst  = (const float*)d_in[5];
    const int* src = eidx;
    const int* dst = eidx + EE;
    float* out = (float*)d_out;

    // ---- workspace carve-out (16B-aligned slices) ------------------------
    char* p = (char*)d_ws;
    __bf16* featb = (__bf16*)p; p += (size_t)NN * IND * sizeof(__bf16);  // 102.4MB
    float*  xp1   = (float*)p;  p += (size_t)NN * HIDD * sizeof(float);  // 102.4MB
    __bf16* h1b   = (__bf16*)p; p += (size_t)NN * HIDD * sizeof(__bf16); // 51.2MB
    __bf16* h2b   = (__bf16*)p; p += (size_t)NN * OUTD * sizeof(__bf16); // 6.4MB
    float*  ebuf  = (float*)p;  p += (size_t)EE * sizeof(float);         // 6.4MB
    float*  asrc  = (float*)p;  p += (size_t)NN * sizeof(float);
    float*  adst  = (float*)p;  p += (size_t)NN * sizeof(float);
    int*    mmax  = (int*)p;    p += (size_t)NN * sizeof(int);
    float*  denom = (float*)p;  p += (size_t)NN * sizeof(float);
    __bf16* w1b   = (__bf16*)p; p += (size_t)IND * HIDD * sizeof(__bf16);
    __bf16* w1tb  = (__bf16*)p; p += (size_t)IND * HIDD * sizeof(__bf16);
    __bf16* w2b   = (__bf16*)p; p += (size_t)HIDD * OUTD * sizeof(__bf16);
    __bf16* w2tb  = (__bf16*)p; p += (size_t)HIDD * OUTD * sizeof(__bf16);
    // agg (f32 scatter target) aliases featb: featb dead after GEMM1, and
    // NN*HIDD*4 == NN*IND*2 bytes exactly.
    float* agg = (float*)featb;

    const dim3 blk(256);
    const int mB   = (NN + 255) / 256;        // 391 row-blocks
    const int nhid = NN * HIDD;

    // ---- precision staging: everything WMMA touches becomes bf16 ---------
    f32_to_bf16<<<dim3((NN * IND / 8 + 255) / 256), blk, 0, stream>>>(
        features, featb, NN * IND / 8);
    f32_to_bf16<<<dim3((IND * HIDD / 8 + 255) / 256), blk, 0, stream>>>(
        W1, w1b, IND * HIDD / 8);                       // W1  [512][256]
    transpose_to_bf16<<<dim3((IND * HIDD + 255) / 256), blk, 0, stream>>>(
        W1, w1tb, IND, HIDD);                           // W1t [256][512]
    f32_to_bf16<<<dim3((HIDD * OUTD / 8 + 255) / 256), blk, 0, stream>>>(
        W2, w2b, HIDD * OUTD / 8);                      // W2  [256][32]
    transpose_to_bf16<<<dim3((HIDD * OUTD + 255) / 256), blk, 0, stream>>>(
        W2, w2tb, HIDD, OUTD);                          // W2t [32][256]

    zero_f32<<<dim3((NN + 255) / 256), blk, 0, stream>>>((float*)mmax, NN);
    zero_f32<<<dim3((NN + 255) / 256), blk, 0, stream>>>(denom, NN);

    // conv1: xp1 = features @ W1           (Bt = W1t)
    gemm_bf16_wmma<<<dim3(mB, HIDD / 32), blk, 0, stream>>>(
        featb, w1tb, xp1, (__bf16*)nullptr, NN, IND, HIDD);

    // per-node attention logits (shared by BOTH GAT layers)
    attn_logits<<<dim3((NN * 32 + 255) / 256), blk, 0, stream>>>(
        xp1, att_src, att_dst, asrc, adst, NN);

    // edge softmax, computed once (conv3 reuses conv1's logits exactly)
    edge_alpha<<<dim3((EE + 255) / 256), blk, 0, stream>>>(
        src, dst, asrc, adst, ebuf, mmax, EE);
    edge_exp<<<dim3((EE + 255) / 256), blk, 0, stream>>>(
        dst, ebuf, mmax, denom, EE);

    // layer-1 propagate: agg = segment_sum(xp1[src] * a);  h1 = elu -> bf16
    zero_f32<<<dim3((nhid + 255) / 256), blk, 0, stream>>>(agg, nhid);
    edge_scatter<<<dim3(EE / 8), blk, 0, stream>>>(
        src, dst, ebuf, denom, xp1, agg, EE);
    elu_to_bf16<<<dim3((nhid / 8 + 255) / 256), blk, 0, stream>>>(
        agg, h1b, nhid / 8);

    // h2 = h1 @ W2 -> d_out (f32) + h2b (bf16 for next GEMM)   (Bt = W2t)
    gemm_bf16_wmma<<<dim3(mB, 1), blk, 0, stream>>>(
        h1b, w2tb, out, h2b, NN, HIDD, OUTD);

    // xp3 = h2 @ W2^T -> xp1 buffer        (Bt = (W2^T)^T = W2)
    gemm_bf16_wmma<<<dim3(mB, HIDD / 32), blk, 0, stream>>>(
        h2b, w2b, xp1, (__bf16*)nullptr, NN, OUTD, HIDD);

    // layer-2 propagate with the SAME attention coefficients
    zero_f32<<<dim3((nhid + 255) / 256), blk, 0, stream>>>(agg, nhid);
    edge_scatter<<<dim3(EE / 8), blk, 0, stream>>>(
        src, dst, ebuf, denom, xp1, agg, EE);
    elu_to_bf16<<<dim3((nhid / 8 + 255) / 256), blk, 0, stream>>>(
        agg, h1b, nhid / 8);                              // h3b reuses h1b

    // h4 = h3 @ W1^T -> second output block  (Bt = (W1^T)^T = W1)
    gemm_bf16_wmma<<<dim3(mB, IND / 32), blk, 0, stream>>>(
        h1b, w1b, out + (size_t)NN * OUTD, (__bf16*)nullptr, NN, HIDD, IND);
}
